// SelfAttention_22462678958488
// MI455X (gfx1250) — matrix operand order repaired and checked by reading, not compiled, not run on hardware
//
#include <hip/hip_runtime.h>

// ---------------- problem constants ----------------
#define BB   2
#define LL   1024
#define CC   2048
#define HH   32
#define HKVN 8
#define DD   64
#define LC   2048
#define KT   (LC + LL)        // 3072 total kv positions
#define NREP (HH / HKVN)      // 4

typedef __attribute__((ext_vector_type(16))) __bf16 bf16x16;
typedef __attribute__((ext_vector_type(8)))  float  f32x8;

union Frag { bf16x16 f; uint4 u[2]; };

__device__ __forceinline__ unsigned short f32_to_bf16(float f) {
    unsigned int u = __float_as_uint(f);
    u += 0x7FFFu + ((u >> 16) & 1u);   // round-to-nearest-even
    return (unsigned short)(u >> 16);
}

// LDS byte offset of a __shared__ object: generic LDS-aperture addresses map to
// LDS via addr[31:0] truncation (CDNA5 ISA 10.2), so the low 32 bits suffice.
__device__ __forceinline__ unsigned lds_off(const void* p) {
    return (unsigned)(unsigned long long)p;
}

// ---------------- fp32 -> bf16 convert ----------------
__global__ void k_cvt_bf16(const float* __restrict__ src,
                           unsigned short* __restrict__ dst, int n) {
    int i = blockIdx.x * blockDim.x + threadIdx.x;
    if (i < n) dst[i] = f32_to_bf16(src[i]);
}

// ---------------- bf16 WMMA GEMM: out[M,N] = A[M,K] * W[N,K]^T ----------------
// A row-major [M,K] bf16, W row-major [N,K] bf16, out fp32 [M,N].
// One wave per 16x16 tile; 8 waves per 256-thread block.
__global__ void k_gemm_bf16(const unsigned short* __restrict__ A,
                            const unsigned short* __restrict__ W,
                            float* __restrict__ out, int M, int N, int K) {
    int lane = threadIdx.x & 31;
    int wave = threadIdx.x >> 5;
    int tile = blockIdx.x * (blockDim.x >> 5) + wave;
    int ntn  = N >> 4;
    int tm = tile / ntn, tn = tile - tm * ntn;
    if (tm >= (M >> 4)) return;
    int half = lane >> 4, l15 = lane & 15;

    // A-fragment: lane row = l15, K elems {half*16 .. half*16+15} contiguous
    const unsigned short* Ar = A + (size_t)(tm * 16 + l15) * K + half * 16;
    // B-fragment: lane col = l15, K elems {half*16 .. half*16+15} contiguous
    const unsigned short* Wr = W + (size_t)(tn * 16 + l15) * K + half * 16;

    f32x8 acc = {};
    for (int k0 = 0; k0 < K; k0 += 32) {
        Frag a, w;
        a.u[0] = *(const uint4*)(Ar + k0);
        a.u[1] = *(const uint4*)(Ar + k0 + 8);
        w.u[0] = *(const uint4*)(Wr + k0);
        w.u[1] = *(const uint4*)(Wr + k0 + 8);
        __builtin_prefetch(Ar + k0 + 128, 0, 1);
        __builtin_prefetch(Wr + k0 + 128, 0, 1);
        acc = __builtin_amdgcn_wmma_f32_16x16x32_bf16(
                  false, a.f, false, w.f, (short)0, acc, false, false);
    }
    // C/D layout: row = r + half*8, col = l15
    float* obase = out + (size_t)(tm * 16) * N + tn * 16;
#pragma unroll
    for (int r = 0; r < 8; ++r)
        obase[(size_t)(r + half * 8) * N + l15] = acc[r];
}

// ---------------- RMSNorm + RoPE, fp32 proj -> bf16 [B, nh, seq, 64] ----------------
// proj: [B, L, nh*64] fp32. One wave per (b,h,l) row of 64.
__global__ void k_norm_rope(const float* __restrict__ proj,
                            const float* __restrict__ nw,
                            const float* __restrict__ cosT,
                            const float* __restrict__ sinT,
                            unsigned short* __restrict__ dst,
                            int nh, int seq_total, int seq_off) {
    int lane = threadIdx.x & 31;
    int wid  = (blockIdx.x * blockDim.x + threadIdx.x) >> 5;
    if (wid >= BB * nh * LL) return;
    int l = wid % LL; int t = wid / LL;
    int h = t % nh;   int b = t / nh;

    const float* row = proj + ((size_t)(b * LL + l) * nh + h) * 64;
    float t1 = row[lane], t2 = row[lane + 32];
    float ss = t1 * t1 + t2 * t2;
#pragma unroll
    for (int off = 16; off >= 1; off >>= 1) ss += __shfl_xor(ss, off, 32);
    float inv = rsqrtf(ss * (1.0f / 64.0f) + 1e-6f);
    float a = t1 * inv * nw[lane];
    float c2 = t2 * inv * nw[lane + 32];
    float c = cosT[l * 32 + lane], s = sinT[l * 32 + lane];
    unsigned short* drow = dst + ((size_t)(b * nh + h) * seq_total + seq_off + l) * 64;
    drow[lane]      = f32_to_bf16(a * c - c2 * s);
    drow[lane + 32] = f32_to_bf16(a * s + c2 * c);
}

// cached_k fp32 [B,HKV,LC,64] -> kb bf16 [B,HKV,KT,64] (positions 0..LC-1)
__global__ void k_cache_k(const float* __restrict__ ck, unsigned short* __restrict__ kb) {
    size_t i = (size_t)blockIdx.x * blockDim.x + threadIdx.x;
    if (i >= (size_t)BB * HKVN * LC * DD) return;
    int d = i & 63; size_t t = i >> 6;
    int lc = t % LC; size_t bh = t / LC;
    kb[(bh * KT + lc) * DD + d] = f32_to_bf16(ck[i]);
}

// cached_v fp32 [B,HKV,LC,64] -> vtb bf16 [B,HKV,64,KT] transposed
__global__ void k_cache_v(const float* __restrict__ cv, unsigned short* __restrict__ vtb) {
    size_t i = (size_t)blockIdx.x * blockDim.x + threadIdx.x;
    if (i >= (size_t)BB * HKVN * LC * DD) return;
    int d = i & 63; size_t t = i >> 6;
    int lc = t % LC; size_t bh = t / LC;
    vtb[(bh * DD + d) * (size_t)KT + lc] = f32_to_bf16(cv[i]);
}

// v_proj fp32 [B,L,HKV*64] -> vtb bf16 [B,HKV,64,KT] at columns LC..KT-1
__global__ void k_vproj_t(const float* __restrict__ vp, unsigned short* __restrict__ vtb) {
    size_t i = (size_t)blockIdx.x * blockDim.x + threadIdx.x;
    if (i >= (size_t)BB * LL * HKVN * DD) return;
    int d = i & 63; size_t t = i >> 6;
    int h = t % HKVN; t /= HKVN;
    int l = t % LL;   int b = (int)(t / LL);
    vtb[((size_t)(b * HKVN + h) * DD + d) * KT + LC + l] = f32_to_bf16(vp[i]);
}

// ---------------- flash attention: one wave per 16-row Q tile ----------------
// K/V chunks are double-buffered into LDS with GLOBAL_LOAD_ASYNC_TO_LDS_B128
// (ASYNCcnt-tracked), hiding HBM/L2 latency behind the previous chunk's WMMAs.
// qb [B,H,L,64] bf16; kb [B,HKV,KT,64] bf16; vtb [B,HKV,64,KT] bf16
// attn_out bf16 [B*L, H*64]
__global__ void k_flash(const unsigned short* __restrict__ qb,
                        const unsigned short* __restrict__ kb,
                        const unsigned short* __restrict__ vtb,
                        unsigned short* __restrict__ attn_out) {
    __shared__ __align__(16) unsigned short klds[2][32 * 64];  // K chunk [kv][d]
    __shared__ __align__(16) unsigned short vlds[2][64 * 32];  // V chunk [d][kv]
    __shared__ __align__(16) unsigned short pbuf[16 * 32];     // P bounce

    int lane = threadIdx.x & 31;
    int half = lane >> 4, l15 = lane & 15;
    int blk = blockIdx.x;
    int qt = blk % (LL / 16); int t = blk / (LL / 16);
    int h = t % HH; int b = t / HH;
    int hkv = h / NREP;

    const unsigned short* Qp = qb  + (size_t)(b * HH   + h)   * LL * DD;
    const unsigned short* Kp = kb  + (size_t)(b * HKVN + hkv) * KT * DD;
    const unsigned short* Vp = vtb + (size_t)(b * HKVN + hkv) * DD * KT;

    // Q A-fragments for both K-steps of the D=64 inner dim (held in VGPRs)
    Frag qa[2];
    int qrow = qt * 16 + l15;
#pragma unroll
    for (int s = 0; s < 2; ++s) {
        const unsigned short* p = Qp + (size_t)qrow * DD + s * 32 + half * 16;
        qa[s].u[0] = *(const uint4*)(p);
        qa[s].u[1] = *(const uint4*)(p + 8);
    }

    // async-stage one 32-kv chunk of K (32x64) and V^T (64x32) into LDS buffer `buf`
    auto issue_chunk = [&](int n0, int buf) {
#pragma unroll
        for (int i = 0; i < 8; ++i) {           // K: 256 x b128
            int e = i * 32 + lane;
            const unsigned short* g = Kp + (size_t)n0 * DD + e * 8;
            unsigned lo = lds_off(&klds[buf][e * 8]);
            asm volatile("global_load_async_to_lds_b128 %0, %1, off"
                         :: "v"(lo), "v"(g) : "memory");
        }
#pragma unroll
        for (int i = 0; i < 8; ++i) {           // V: 256 x b128 (4 per head-dim row)
            int e = i * 32 + lane;
            int d = e >> 2, seg = e & 3;
            const unsigned short* g = Vp + (size_t)d * KT + n0 + seg * 8;
            unsigned lo = lds_off(&vlds[buf][d * 32 + seg * 8]);
            asm volatile("global_load_async_to_lds_b128 %0, %1, off"
                         :: "v"(lo), "v"(g) : "memory");
        }
    };

    f32x8 Oacc[4] = {{}, {}, {}, {}};
    float mrun[8], lrun[8];
#pragma unroll
    for (int r = 0; r < 8; ++r) { mrun[r] = -1e30f; lrun[r] = 0.0f; }

    const float scale = 0.125f;   // D^-0.5

    issue_chunk(0, 0);

    for (int c = 0; c < KT / 32; ++c) {
        int n0 = c * 32;
        int cur = c & 1;
        if (c + 1 < KT / 32) {
            // DS reads of the other buffer (from chunk c-1) must finish before
            // async writes can reuse it: DS and ASYNC LDS traffic are unordered.
            asm volatile("s_wait_dscnt 0" ::: "memory");
            issue_chunk(n0 + 32, cur ^ 1);
            asm volatile("s_wait_asynccnt 16" ::: "memory");  // chunk c landed
        } else {
            asm volatile("s_wait_asynccnt 0" ::: "memory");
        }
        const unsigned short* Kl = klds[cur];
        const unsigned short* Vl = vlds[cur];

        // ---- S = Q * K^T for two 16-col subtiles (K frags from LDS) ----
        f32x8 S[2];
#pragma unroll
        for (int st = 0; st < 2; ++st) {
            f32x8 s = {};
            int rl = st * 16 + l15;
#pragma unroll
            for (int ks = 0; ks < 2; ++ks) {
                Frag kf;
                const unsigned short* p = Kl + rl * DD + ks * 32 + half * 16;
                kf.u[0] = *(const uint4*)(p);
                kf.u[1] = *(const uint4*)(p + 8);
                s = __builtin_amdgcn_wmma_f32_16x16x32_bf16(
                        false, qa[ks].f, false, kf.f, (short)0, s, false, false);
            }
            S[st] = s;
        }

        // ---- online softmax (row = r + half*8, cols across 16 lanes of a half) ----
        float p0[8], p1[8], sf[8];
#pragma unroll
        for (int r = 0; r < 8; ++r) {
            float s0 = S[0][r] * scale, s1 = S[1][r] * scale;
            float cm = fmaxf(s0, s1);
#pragma unroll
            for (int off = 8; off >= 1; off >>= 1)
                cm = fmaxf(cm, __shfl_xor(cm, off, 32));
            float nm = fmaxf(mrun[r], cm);
            float f  = __expf(mrun[r] - nm);
            float e0 = __expf(s0 - nm), e1 = __expf(s1 - nm);
            float rs = e0 + e1;
#pragma unroll
            for (int off = 8; off >= 1; off >>= 1)
                rs += __shfl_xor(rs, off, 32);
            lrun[r] = lrun[r] * f + rs;
            mrun[r] = nm;
            sf[r] = f; p0[r] = e0; p1[r] = e1;
        }
#pragma unroll
        for (int nt = 0; nt < 4; ++nt)
#pragma unroll
            for (int r = 0; r < 8; ++r) Oacc[nt][r] *= sf[r];

        // ---- P (16x32) through LDS to re-layout as A-fragment ----
#pragma unroll
        for (int r = 0; r < 8; ++r) {
            int row = r + half * 8;
            pbuf[row * 32 + l15]      = f32_to_bf16(p0[r]);
            pbuf[row * 32 + 16 + l15] = f32_to_bf16(p1[r]);
        }
        asm volatile("s_wait_dscnt 0" ::: "memory");
        Frag pa;
        pa.u[0] = *(const uint4*)(pbuf + l15 * 32 + half * 16);
        pa.u[1] = *(const uint4*)(pbuf + l15 * 32 + half * 16 + 8);

        // ---- O += P * V (V frags from LDS, contiguous) ----
#pragma unroll
        for (int nt = 0; nt < 4; ++nt) {
            int d = nt * 16 + l15;
            Frag vf;
            const unsigned short* p = Vl + d * 32 + half * 16;
            vf.u[0] = *(const uint4*)(p);
            vf.u[1] = *(const uint4*)(p + 8);
            Oacc[nt] = __builtin_amdgcn_wmma_f32_16x16x32_bf16(
                           false, pa.f, false, vf.f, (short)0, Oacc[nt], false, false);
        }
    }

    // ---- finalize: O /= l, write bf16 [B*L, H*64] ----
#pragma unroll
    for (int nt = 0; nt < 4; ++nt)
#pragma unroll
        for (int r = 0; r < 8; ++r) {
            int row = qt * 16 + r + half * 8;
            float o = Oacc[nt][r] / lrun[r];
            attn_out[(size_t)(b * LL + row) * (HH * DD) + h * 64 + nt * 16 + l15] =
                f32_to_bf16(o);
        }
}

// ---------------- host launch ----------------
extern "C" void kernel_launch(void* const* d_in, const int* in_sizes, int n_in,
                              void* d_out, int out_size, void* d_ws, size_t ws_size,
                              hipStream_t stream) {
    (void)in_sizes; (void)n_in; (void)out_size; (void)ws_size;
    const float* x        = (const float*)d_in[0];
    const float* cached_k = (const float*)d_in[1];
    const float* cached_v = (const float*)d_in[2];
    const float* rope_cos = (const float*)d_in[3];
    const float* rope_sin = (const float*)d_in[4];
    const float* Wq       = (const float*)d_in[5];
    const float* Wk       = (const float*)d_in[6];
    const float* Wv       = (const float*)d_in[7];
    const float* Wo       = (const float*)d_in[8];
    const float* q_norm_w = (const float*)d_in[9];
    const float* k_norm_w = (const float*)d_in[10];
    float* out = (float*)d_out;

    // workspace carve-up (256B aligned)
    char* wp = (char*)d_ws;
    auto carve = [&](size_t bytes) {
        void* p = (void*)wp;
        wp += (bytes + 255) & ~(size_t)255;
        return p;
    };
    const size_t M = (size_t)BB * LL;                       // 2048
    unsigned short* xb   = (unsigned short*)carve(M * CC * 2);
    unsigned short* Wqb  = (unsigned short*)carve((size_t)HH * DD * CC * 2);
    unsigned short* Wkb  = (unsigned short*)carve((size_t)HKVN * DD * CC * 2);
    unsigned short* Wvb  = (unsigned short*)carve((size_t)HKVN * DD * CC * 2);
    unsigned short* Wob  = (unsigned short*)carve((size_t)CC * HH * DD * 2);
    float* qproj         = (float*)carve(M * (HH * DD) * 4);
    float* kproj         = (float*)carve(M * (HKVN * DD) * 4);
    float* vproj         = (float*)carve(M * (HKVN * DD) * 4);
    unsigned short* qb   = (unsigned short*)carve((size_t)BB * HH * LL * DD * 2);
    unsigned short* kbuf = (unsigned short*)carve((size_t)BB * HKVN * KT * DD * 2);
    unsigned short* vtb  = (unsigned short*)carve((size_t)BB * HKVN * DD * KT * 2);
    unsigned short* attn = (unsigned short*)carve(M * (HH * DD) * 2);

    auto blocks = [](size_t n, int t) { return (unsigned)((n + t - 1) / t); };

    // 1) fp32 -> bf16 conversions
    k_cvt_bf16<<<blocks(M * CC, 256), 256, 0, stream>>>(x, xb, (int)(M * CC));
    k_cvt_bf16<<<blocks((size_t)HH * DD * CC, 256), 256, 0, stream>>>(Wq, Wqb, HH * DD * CC);
    k_cvt_bf16<<<blocks((size_t)HKVN * DD * CC, 256), 256, 0, stream>>>(Wk, Wkb, HKVN * DD * CC);
    k_cvt_bf16<<<blocks((size_t)HKVN * DD * CC, 256), 256, 0, stream>>>(Wv, Wvb, HKVN * DD * CC);
    k_cvt_bf16<<<blocks((size_t)CC * HH * DD, 256), 256, 0, stream>>>(Wo, Wob, CC * HH * DD);

    // 2) QKV projection GEMMs (WMMA bf16, fp32 accumulate)
    k_gemm_bf16<<<(2048 / 16) * (2048 / 16) / 8, 256, 0, stream>>>(xb, Wqb, qproj, 2048, 2048, 2048);
    k_gemm_bf16<<<(2048 / 16) * (512 / 16) / 8, 256, 0, stream>>>(xb, Wkb, kproj, 2048, 512, 2048);
    k_gemm_bf16<<<(2048 / 16) * (512 / 16) / 8, 256, 0, stream>>>(xb, Wvb, vproj, 2048, 512, 2048);

    // 3) RMSNorm + RoPE (q, k) and KV cache assembly (k append, v transpose)
    k_norm_rope<<<blocks((size_t)BB * HH * LL * 32, 256), 256, 0, stream>>>(
        qproj, q_norm_w, rope_cos, rope_sin, qb, HH, LL, 0);
    k_norm_rope<<<blocks((size_t)BB * HKVN * LL * 32, 256), 256, 0, stream>>>(
        kproj, k_norm_w, rope_cos, rope_sin, kbuf, HKVN, KT, LC);
    k_cache_k<<<blocks((size_t)BB * HKVN * LC * DD, 256), 256, 0, stream>>>(cached_k, kbuf);
    k_cache_v<<<blocks((size_t)BB * HKVN * LC * DD, 256), 256, 0, stream>>>(cached_v, vtb);
    k_vproj_t<<<blocks((size_t)BB * LL * HKVN * DD, 256), 256, 0, stream>>>(vproj, vtb);

    // 4) fused flash attention (WMMA S and PV, async double-buffered K/V in LDS)
    k_flash<<<BB * HH * (LL / 16), 32, 0, stream>>>(qb, kbuf, vtb, attn);

    // 5) output projection -> fp32 d_out
    k_gemm_bf16<<<(2048 / 16) * (2048 / 16) / 8, 256, 0, stream>>>(attn, Wob, out, 2048, 2048, 2048);
}
